// Net_4200478015619
// MI455X (gfx1250) — matrix operand order repaired and checked by reading, not compiled, not run on hardware
//
#include <hip/hip_runtime.h>

typedef __attribute__((ext_vector_type(16))) _Float16 v16h;
typedef __attribute__((ext_vector_type(8)))  float    v8f;

#define KDIM     128
#define M_HEADS  100
#define MT       7            // 7 column tiles of 16 -> 112 (zero padded)
#define W_STRIDE 144          // LDS row stride in halfs (bank-rotating pad)
#define ALPHA    0.1f

// Load 16 contiguous f32 (K = base..base+15) as the A fragment and split into
// hi/lo f16 (x = hi + lo, |err| ~ 2^-22 relative after 3-product WMMA).
__device__ __forceinline__ void load_split_a(const float* __restrict__ p,
                                             int base, v16h& h, v16h& l) {
#pragma unroll
  for (int c = 0; c < 2; ++c) {          // K chunks [base, base+8) and [base+8, base+16)
    const float4 f0 = *(const float4*)(p + base + c * 8);
    const float4 f1 = *(const float4*)(p + base + c * 8 + 4);
    const float fv[8] = {f0.x, f0.y, f0.z, f0.w, f1.x, f1.y, f1.z, f1.w};
#pragma unroll
    for (int e = 0; e < 8; ++e) {
      const _Float16 hh = (_Float16)fv[e];
      h[c * 8 + e] = hh;
      l[c * 8 + e] = (_Float16)(fv[e] - (float)hh);
    }
  }
}

// 16 rows/wave: true demand ~150 VGPRs. The per-iteration memory barrier in
// the loop stops LICM from hoisting the (rb-invariant) LDS B-fragments out of
// the grid-stride loop -- that hoist is what forced scratch spills before.
__global__ __launch_bounds__(256, 4) void fused_heads_kernel(
    const float* __restrict__ x, const float* __restrict__ W,
    float* __restrict__ out, int rowBlocksTotal, int wavesTotal)
{
  __shared__ _Float16 Wh[112 * W_STRIDE];
  __shared__ _Float16 Wl[112 * W_STRIDE];

  const int tid = threadIdx.x;

  // Stage W (100x128 f32) into LDS as hi/lo f16 split, zero-padded to 112 rows.
  for (int i = tid; i < 112 * KDIM; i += 256) {
    const int n = i >> 7;
    const int k = i & (KDIM - 1);
    const float w = (n < M_HEADS) ? W[n * KDIM + k] : 0.0f;
    const _Float16 wh = (_Float16)w;
    Wh[n * W_STRIDE + k] = wh;
    Wl[n * W_STRIDE + k] = (_Float16)(w - (float)wh);
  }
  __syncthreads();

  const int lane   = tid & 31;
  const int waveGl = blockIdx.x * 8 + (tid >> 5);
  const int half16 = lane >> 4;   // 0: lanes 0-15, 1: lanes 16-31
  const int l16    = lane & 15;

  for (int rb = waveGl; rb < rowBlocksTotal; rb += wavesTotal) {
    // Keep all memory loads (notably the LDS B-fragments) inside the loop:
    // without this, LICM hoists 56 x 8 VGPRs of fragments and spills them.
    asm volatile("" ::: "memory");

    const int rowBase = rb * 16;
    const float* __restrict__ xrow = x + (size_t)(rowBase + l16) * KDIM;

    // Prefetch this lane's row for the next grid-stride tile.
    {
      const int nrb = rb + wavesTotal;
      if (nrb < rowBlocksTotal)
        __builtin_prefetch((const void*)(x + (size_t)(nrb * 16 + l16) * KDIM), 0, 1);
    }

    v8f acc[MT];
    const v8f vzero = {};
#pragma unroll
    for (int t = 0; t < MT; ++t) acc[t] = vzero;

#pragma unroll
    for (int kk = 0; kk < 4; ++kk) {
      // A fragments for this 32-wide K step only (small live range).
      const int abase = kk * 32 + half16 * 16;
      v16h ah, al;
      load_split_a(xrow, abase, ah, al);

      const int kb = kk * 32 + half16 * 16;
#pragma unroll
      for (int t = 0; t < MT; ++t) {
        const int n = t * 16 + l16;   // head column this lane supplies for B
        const v16h bh = *(const v16h*)&Wh[n * W_STRIDE + kb];
        const v16h bl = *(const v16h*)&Wl[n * W_STRIDE + kb];
        acc[t] = __builtin_amdgcn_wmma_f32_16x16x32_f16(
            false, ah, false, bh, (short)0, acc[t], false, false);
        acc[t] = __builtin_amdgcn_wmma_f32_16x16x32_f16(
            false, ah, false, bl, (short)0, acc[t], false, false);
        acc[t] = __builtin_amdgcn_wmma_f32_16x16x32_f16(
            false, al, false, bh, (short)0, acc[t], false, false);
      }
    }

    // --- Cubic nonlinearity + per-lane partial sum over head tiles ---
    // C layout: VGPR j holds x-row rowBase + j + 8*half16, column = l16.
    float s[8];
#pragma unroll
    for (int j = 0; j < 8; ++j) {
      float a = 0.0f;
#pragma unroll
      for (int t = 0; t < MT; ++t) {
        const float z = acc[t][j];
        a += z * z * z + ALPHA * z;
      }
      s[j] = a;
    }

    // Reduce across the 16 column-lanes (stays inside each 16-lane half).
#pragma unroll
    for (int off = 8; off >= 1; off >>= 1) {
#pragma unroll
      for (int j = 0; j < 8; ++j)
        s[j] += __shfl_xor(s[j], off, 32);
    }

    if (l16 == 0) {
      const int r0 = rowBase + half16 * 8;
#pragma unroll
      for (int j = 0; j < 8; ++j) out[r0 + j] = s[j];
    }
  }
}

extern "C" void kernel_launch(void* const* d_in, const int* in_sizes, int n_in,
                              void* d_out, int out_size, void* d_ws, size_t ws_size,
                              hipStream_t stream) {
  (void)n_in; (void)d_ws; (void)ws_size; (void)out_size;
  const float* x = (const float*)d_in[0];
  const float* W = (const float*)d_in[1];
  float* out = (float*)d_out;

  const int Brows     = in_sizes[0] / KDIM;   // 262144
  const int rowBlocks = Brows / 16;           // 16384
  const int blocks    = 1024;                 // 8 waves each -> 8192 waves, 2 iters
  const int wavesTot  = blocks * 8;

  fused_heads_kernel<<<blocks, 256, 0, stream>>>(x, W, out, rowBlocks, wavesTot);
}
